// AttentionNet_61014305407294
// MI455X (gfx1250) — matrix operand order repaired and checked by reading, not compiled, not run on hardware
//
#include <hip/hip_runtime.h>
#include <hip/hip_bf16.h>

typedef __attribute__((ext_vector_type(16))) __bf16 v16bf;
typedef __attribute__((ext_vector_type(8)))  __bf16 v8bf;
typedef __attribute__((ext_vector_type(8)))  float  v8f;

#define DIM    512
#define MLP    768
#define NSEQ   1024
#define BATCH  8
#define MROWS  (BATCH * NSEQ)     // 8192
#define KKSEL  716                // int(1024 * 0.7)
#define SCALE  0.125f             // 64^-0.5

// ---------------------------------------------------------------------------
// WMMA fragment helpers (V_WMMA_F32_16X16X32_BF16; A/B k order is this code's contiguous choice):
// A 16x32 bf16 : lanes 0-15 -> M=lane, K = h*16..h*16+15
// B 32x16 bf16 : lanes 0-15 -> N=lane, K = h*16..h*16+15   (BT[N][K] rows)
// C/D f32 16x16: lane%16 = N, VGPR r -> M = (lane/16)*8 + r
// ---------------------------------------------------------------------------
__device__ __forceinline__ v16bf load_afrag(const __bf16* Arow, int k0, int half) {
  return *(const v16bf*)(Arow + k0 + half * 16);
}
__device__ __forceinline__ v16bf load_bfrag(const __bf16* Brow, int k0, int half) {
  return *(const v16bf*)(Brow + k0 + half * 16);
}
__device__ __forceinline__ v8f wmma_bf16(v16bf a, v16bf b, v8f c) {
  return __builtin_amdgcn_wmma_f32_16x16x32_bf16(false, a, false, b,
                                                 (short)0, c, false, false);
}

// single-tile accumulate (used by the attn@V k-split partials)
__device__ __forceinline__ v8f wmma_accum(const __bf16* __restrict__ Arow,
                                          const __bf16* __restrict__ Brow,
                                          int K, int half, v8f acc) {
  for (int k0 = 0; k0 < K; k0 += 32) {
    __builtin_prefetch((const void*)(Brow + k0 + 128), 0, 1);  // speculative
    acc = wmma_bf16(load_afrag(Arow, k0, half), load_bfrag(Brow, k0, half), acc);
  }
  return acc;
}

// 2x4 register-blocked mainloop: one wave -> 32x64 output, 8 wmma per k-chunk,
// 12 b128 loads per 8 wmma (vs 32 for unblocked).
__device__ __forceinline__ void wmma_block_2x4(const __bf16* A, int lda, int m0,
                                               const __bf16* BT, int ldb, int n0,
                                               int K, v8f acc[2][4]) {
  int lane = threadIdx.x & 31;
  int r    = lane & 15;
  int half = lane >> 4;
  const __bf16* Ar[2];
  const __bf16* Br[4];
#pragma unroll
  for (int i = 0; i < 2; ++i) Ar[i] = A + (long)(m0 + i * 16 + r) * lda;
#pragma unroll
  for (int j = 0; j < 4; ++j) Br[j] = BT + (long)(n0 + j * 16 + r) * ldb;

  for (int k0 = 0; k0 < K; k0 += 32) {
    // unconditional speculative prefetch: no branch / hazard NOPs in mainloop
    __builtin_prefetch((const void*)(Ar[0] + k0 + 128), 0, 1);
    __builtin_prefetch((const void*)(Ar[1] + k0 + 128), 0, 1);
    v16bf a[2], b[4];
#pragma unroll
    for (int i = 0; i < 2; ++i) a[i] = load_afrag(Ar[i], k0, half);
#pragma unroll
    for (int j = 0; j < 4; ++j) b[j] = load_bfrag(Br[j], k0, half);
#pragma unroll
    for (int i = 0; i < 2; ++i)
#pragma unroll
      for (int j = 0; j < 4; ++j)
        acc[i][j] = wmma_bf16(a[i], b[j], acc[i][j]);
  }
}

__device__ __forceinline__ void zero_acc(v8f acc[2][4]) {
#pragma unroll
  for (int i = 0; i < 2; ++i)
#pragma unroll
    for (int j = 0; j < 4; ++j)
#pragma unroll
      for (int e = 0; e < 8; ++e) acc[i][j][e] = 0.f;
}

// ---------------------------------------------------------------------------
// Stage 1: dtype conversions / weight transposes
// ---------------------------------------------------------------------------
__global__ __launch_bounds__(256) void cvt_bf16_kernel(const float* __restrict__ src,
                                                       __bf16* __restrict__ dst, long n) {
  long i = (long)blockIdx.x * blockDim.x + threadIdx.x;
  if (i < n) dst[i] = (__bf16)src[i];
}

__global__ __launch_bounds__(256) void transpose_bf16_kernel(const float* __restrict__ src,
                                                             __bf16* __restrict__ dst,
                                                             int R, int C) {
  long i = (long)blockIdx.x * blockDim.x + threadIdx.x;
  if (i < (long)R * C) {
    int r = (int)(i / C), c = (int)(i % C);
    dst[(long)c * R + r] = (__bf16)src[i];
  }
}

// ---------------------------------------------------------------------------
// Stage 2: QKV projections.  z=0: Q=x_r@Wq  z=1: K=x_d@Wk  z=2: V=x_r@Wv
// Q,K stored [b,h,n,d] bf16 ; V stored transposed [b,h,d,n] bf16
// Block = 8 waves -> 64x256 tile; wave(wm,wn) -> 32x64 sub-tile.
// ---------------------------------------------------------------------------
__global__ __launch_bounds__(256) void qkv_gemm_kernel(
    const __bf16* __restrict__ xr, const __bf16* __restrict__ xd,
    const __bf16* __restrict__ WqT, const __bf16* __restrict__ WkT,
    const __bf16* __restrict__ WvT,
    __bf16* __restrict__ Qb, __bf16* __restrict__ Kb, __bf16* __restrict__ Vt) {
  int wave = threadIdx.x >> 5;
  int lane = threadIdx.x & 31;
  int wm = wave >> 2, wn = wave & 3;
  int m0 = blockIdx.x * 64 + wm * 32;
  int n0 = blockIdx.y * 256 + wn * 64;
  int z  = blockIdx.z;

  const __bf16* X  = (z == 1) ? xd : xr;
  const __bf16* WT = (z == 0) ? WqT : (z == 1) ? WkT : WvT;

  v8f acc[2][4];
  zero_acc(acc);
  wmma_block_2x4(X, DIM, m0, WT, DIM, n0, DIM, acc);

  int col  = lane & 15;
  int row0 = (lane >> 4) * 8;
#pragma unroll
  for (int i = 0; i < 2; ++i)
#pragma unroll
    for (int j = 0; j < 4; ++j) {
      int n = n0 + j * 16 + col;
      int h = n >> 6, d = n & 63;
#pragma unroll
      for (int rr = 0; rr < 8; ++rr) {
        int m  = m0 + i * 16 + row0 + rr;
        int bb = m >> 10, nn = m & (NSEQ - 1);
        __bf16 v = (__bf16)acc[i][j][rr];
        long bhn = (long)(bb * 8 + h);
        if (z == 0)      Qb[(bhn * NSEQ + nn) * 64 + d] = v;
        else if (z == 1) Kb[(bhn * NSEQ + nn) * 64 + d] = v;
        else             Vt[(bhn * 64 + d) * NSEQ + nn] = v;
      }
    }
}

// ---------------------------------------------------------------------------
// Stage 3: attention with exact top-k(716) row threshold + softmax + attn@V
// One workgroup (8 waves) handles 16 query rows of one (b,h).
// ---------------------------------------------------------------------------
__global__ __launch_bounds__(256) void attn_kernel(
    const __bf16* __restrict__ Qb, const __bf16* __restrict__ Kb,
    const __bf16* __restrict__ Vt, float* __restrict__ attn_out) {
  __shared__ __align__(32) float sd[16 * NSEQ];   // 64 KB dots row-block

  int wave = threadIdx.x >> 5;
  int lane = threadIdx.x & 31;
  int bh = blockIdx.x >> 6;            // 64 row-tiles per (b,h)
  int mt = blockIdx.x & 63;
  int m0 = mt * 16;

  const __bf16* Qbase = Qb + (long)bh * NSEQ * 64 + (long)m0 * 64;
  const __bf16* Kbase = Kb + (long)bh * NSEQ * 64;
  const __bf16* Vbase = Vt + (long)bh * 64 * NSEQ;

  int col  = lane & 15;
  int row0 = (lane >> 4) * 8;
  int r    = lane & 15;
  int half = lane >> 4;

  // ---- dots = (Q_tile @ K^T) * SCALE  -> LDS -------------------------------
  // Q A-fragments (K=64 -> 2 chunks) hoisted: loaded once, reused for this wave's 8 tiles
  {
    const __bf16* Arow = Qbase + (long)r * 64;
    v16bf a0 = load_afrag(Arow, 0, half);
    v16bf a1 = load_afrag(Arow, 32, half);
    for (int t = 0; t < 8; ++t) {
      int j0 = (t * 8 + wave) * 16;
      const __bf16* Brow = Kbase + (long)(j0 + r) * 64;
      v8f acc = {0.f, 0.f, 0.f, 0.f, 0.f, 0.f, 0.f, 0.f};
      acc = wmma_bf16(a0, load_bfrag(Brow, 0, half), acc);
      acc = wmma_bf16(a1, load_bfrag(Brow, 32, half), acc);
#pragma unroll
      for (int rr = 0; rr < 8; ++rr)
        sd[(row0 + rr) * NSEQ + j0 + col] = acc[rr] * SCALE;
    }
  }
  __syncthreads();

  // ---- per-row: exact k-th-largest via radix select, masked softmax --------
  __bf16* abf = (__bf16*)sd;           // bf16 attn written in-place (row-local)
  for (int ii = 0; ii < 2; ++ii) {
    int i = wave + ii * 8;             // this wave owns rows {wave, wave+8}
    float    val[32];
    unsigned key[32];
#pragma unroll
    for (int t = 0; t < 32; ++t) {
      float f = sd[i * NSEQ + t * 32 + lane];
      val[t] = f;
      unsigned u = __float_as_uint(f);
      key[t] = (u & 0x80000000u) ? ~u : (u | 0x80000000u);  // order-preserving
    }
    // max T with count(key >= T) >= KKSEL  ==  sortable key of k-th largest
    unsigned T = 0u;
    for (int bit = 31; bit >= 0; --bit) {
      unsigned cand = T | (1u << bit);
      int c = 0;
#pragma unroll
      for (int t = 0; t < 32; ++t) c += (key[t] >= cand) ? 1 : 0;
      for (int off = 16; off; off >>= 1) c += __shfl_xor(c, off, 32);
      if (c >= KKSEL) T = cand;
    }
    float mx = -3.402823466e38f;
#pragma unroll
    for (int t = 0; t < 32; ++t) mx = fmaxf(mx, val[t]);
    for (int off = 16; off; off >>= 1) mx = fmaxf(mx, __shfl_xor(mx, off, 32));
    float s = 0.f;
#pragma unroll
    for (int t = 0; t < 32; ++t)
      if (key[t] >= T) s += __expf(val[t] - mx);
    for (int off = 16; off; off >>= 1) s += __shfl_xor(s, off, 32);
    float inv = 1.0f / s;
#pragma unroll
    for (int t = 0; t < 32; ++t) {
      float p = (key[t] >= T) ? __expf(val[t] - mx) * inv : 0.f;
      abf[i * 2048 + t * 32 + lane] = (__bf16)p;   // stride 4KB: stays in row's span
    }
  }
  __syncthreads();

  // ---- out_tile = attn(16x1024) @ V(1024x64), all 8 waves via k-split -----
  {
    int ntile = wave & 3;              // d-tile
    int kh    = wave >> 2;             // k half: 0 -> [0,512), 1 -> [512,1024)
    const __bf16* Arow = abf + (long)r * 2048 + kh * 512;
    const __bf16* Brow = Vbase + (long)(ntile * 16 + r) * NSEQ + kh * 512;
    v8f acc = {0.f, 0.f, 0.f, 0.f, 0.f, 0.f, 0.f, 0.f};
    acc = wmma_accum(Arow, Brow, 512, half, acc);
    __syncthreads();                   // attn reads done; sd reusable
    float* sp = sd;                    // 4 tiles x 256 floats = 4 KB scratch
    if (kh == 1) {
#pragma unroll
      for (int rr = 0; rr < 8; ++rr) sp[ntile * 256 + lane * 8 + rr] = acc[rr];
    }
    __syncthreads();
    if (kh == 0) {
      int b = bh >> 3, h = bh & 7;
#pragma unroll
      for (int rr = 0; rr < 8; ++rr) {
        float v = acc[rr] + sp[ntile * 256 + lane * 8 + rr];
        int m = m0 + row0 + rr;
        int c = h * 64 + ntile * 16 + col;
        attn_out[((long)b * NSEQ + m) * DIM + c] = v;
      }
    }
  }
}

// ---------------------------------------------------------------------------
// Stage 4: out = LN(attn_out)*g+b + x_r  (-> d_out, the residual stream)
//          h   = LN(out)*fg+fb          (-> bf16 FFN input)
// ---------------------------------------------------------------------------
__global__ __launch_bounds__(256) void ln_kernel(
    const float* __restrict__ attn_out, const float* __restrict__ xr,
    const float* __restrict__ g, const float* __restrict__ b,
    const float* __restrict__ fg, const float* __restrict__ fb,
    float* __restrict__ out_res, __bf16* __restrict__ hbf) {
  __shared__ float sy[DIM];
  __shared__ float rs[256], rq[256];
  int row = blockIdx.x, tid = threadIdx.x;
  const float* x = attn_out + (long)row * DIM;

  float x0 = x[tid], x1 = x[tid + 256];
  rs[tid] = x0 + x1;
  rq[tid] = x0 * x0 + x1 * x1;
  __syncthreads();
  for (int off = 128; off; off >>= 1) {
    if (tid < off) { rs[tid] += rs[tid + off]; rq[tid] += rq[tid + off]; }
    __syncthreads();
  }
  float mu   = rs[0] * (1.f / DIM);
  float var  = rq[0] * (1.f / DIM) - mu * mu;
  float rstd = rsqrtf(var + 1e-5f);

  for (int c = tid; c < DIM; c += 256) {
    float y = (x[c] - mu) * rstd * g[c] + b[c] + xr[(long)row * DIM + c];
    sy[c] = y;
    out_res[(long)row * DIM + c] = y;
  }
  __syncthreads();

  float y0 = sy[tid], y1 = sy[tid + 256];
  rs[tid] = y0 + y1;
  rq[tid] = y0 * y0 + y1 * y1;
  __syncthreads();
  for (int off = 128; off; off >>= 1) {
    if (tid < off) { rs[tid] += rs[tid + off]; rq[tid] += rq[tid + off]; }
    __syncthreads();
  }
  float mu2   = rs[0] * (1.f / DIM);
  float var2  = rq[0] * (1.f / DIM) - mu2 * mu2;
  float rstd2 = rsqrtf(var2 + 1e-5f);

  for (int c = tid; c < DIM; c += 256)
    hbf[(long)row * DIM + c] = (__bf16)((sy[c] - mu2) * rstd2 * fg[c] + fb[c]);
}

// ---------------------------------------------------------------------------
// Stage 5: H1 = gelu_exact(h @ W1 + b1)   [8192 x 768] bf16
// ---------------------------------------------------------------------------
__global__ __launch_bounds__(256) void ffn1_kernel(
    const __bf16* __restrict__ hbf, const __bf16* __restrict__ W1T,
    const float* __restrict__ b1, __bf16* __restrict__ H1) {
  int wave = threadIdx.x >> 5, lane = threadIdx.x & 31;
  int wm = wave >> 2, wn = wave & 3;
  int m0 = blockIdx.x * 64 + wm * 32;
  int n0 = blockIdx.y * 256 + wn * 64;

  v8f acc[2][4];
  zero_acc(acc);
  wmma_block_2x4(hbf, DIM, m0, W1T, DIM, n0, DIM, acc);

  int col = lane & 15, row0 = (lane >> 4) * 8;
#pragma unroll
  for (int i = 0; i < 2; ++i)
#pragma unroll
    for (int j = 0; j < 4; ++j) {
      int n = n0 + j * 16 + col;
      float bias = b1[n];
#pragma unroll
      for (int rr = 0; rr < 8; ++rr) {
        float xv = acc[i][j][rr] + bias;
        float gl = 0.5f * xv * (1.0f + erff(xv * 0.70710678118654752f));
        H1[(long)(m0 + i * 16 + row0 + rr) * MLP + n] = (__bf16)gl;
      }
    }
}

// ---------------------------------------------------------------------------
// Stage 6: final = H1 @ W2 + b2 + out_res   (read-modify-write d_out)
// ---------------------------------------------------------------------------
__global__ __launch_bounds__(256) void ffn2_kernel(
    const __bf16* __restrict__ H1, const __bf16* __restrict__ W2T,
    const float* __restrict__ b2, float* __restrict__ out) {
  int wave = threadIdx.x >> 5, lane = threadIdx.x & 31;
  int wm = wave >> 2, wn = wave & 3;
  int m0 = blockIdx.x * 64 + wm * 32;
  int n0 = blockIdx.y * 256 + wn * 64;

  v8f acc[2][4];
  zero_acc(acc);
  wmma_block_2x4(H1, MLP, m0, W2T, MLP, n0, MLP, acc);

  int col = lane & 15, row0 = (lane >> 4) * 8;
#pragma unroll
  for (int i = 0; i < 2; ++i)
#pragma unroll
    for (int j = 0; j < 4; ++j) {
      int n = n0 + j * 16 + col;
      float bias = b2[n];
#pragma unroll
      for (int rr = 0; rr < 8; ++rr) {
        long idx = (long)(m0 + i * 16 + row0 + rr) * DIM + n;
        out[idx] = acc[i][j][rr] + bias + out[idx];
      }
    }
}

// ---------------------------------------------------------------------------
extern "C" void kernel_launch(void* const* d_in, const int* in_sizes, int n_in,
                              void* d_out, int out_size, void* d_ws, size_t ws_size,
                              hipStream_t stream) {
  const float* x_r  = (const float*)d_in[0];
  const float* x_d  = (const float*)d_in[1];
  const float* Wq   = (const float*)d_in[2];
  const float* Wk   = (const float*)d_in[3];
  const float* Wv   = (const float*)d_in[4];
  const float* ln_g = (const float*)d_in[5];
  const float* ln_b = (const float*)d_in[6];
  const float* fg   = (const float*)d_in[7];
  const float* fb   = (const float*)d_in[8];
  const float* W1   = (const float*)d_in[9];
  const float* b1   = (const float*)d_in[10];
  const float* W2   = (const float*)d_in[11];
  const float* b2   = (const float*)d_in[12];
  float* out = (float*)d_out;

  char* ws = (char*)d_ws;
  auto alloc = [&](size_t bytes) {
    char* p = ws;
    ws += (bytes + 255) & ~(size_t)255;
    return p;
  };
  __bf16* xr_bf = (__bf16*)alloc((size_t)MROWS * DIM * 2);
  __bf16* xd_bf = (__bf16*)alloc((size_t)MROWS * DIM * 2);
  __bf16* WqT   = (__bf16*)alloc((size_t)DIM * DIM * 2);
  __bf16* WkT   = (__bf16*)alloc((size_t)DIM * DIM * 2);
  __bf16* WvT   = (__bf16*)alloc((size_t)DIM * DIM * 2);
  __bf16* W1T   = (__bf16*)alloc((size_t)MLP * DIM * 2);
  __bf16* W2T   = (__bf16*)alloc((size_t)DIM * MLP * 2);
  __bf16* Qb    = (__bf16*)alloc((size_t)MROWS * DIM * 2);   // [b,h,n,d]
  __bf16* Kb    = (__bf16*)alloc((size_t)MROWS * DIM * 2);   // [b,h,n,d]
  __bf16* Vt    = (__bf16*)alloc((size_t)MROWS * DIM * 2);   // [b,h,d,n]
  float*  attnO = (float*) alloc((size_t)MROWS * DIM * 4);
  __bf16* hbf   = (__bf16*)alloc((size_t)MROWS * DIM * 2);
  __bf16* H1    = (__bf16*)alloc((size_t)MROWS * MLP * 2);
  (void)in_sizes; (void)n_in; (void)out_size; (void)ws_size;

  long nact = (long)MROWS * DIM;                       // 4,194,304
  cvt_bf16_kernel<<<(int)((nact + 255) / 256), 256, 0, stream>>>(x_r, xr_bf, nact);
  cvt_bf16_kernel<<<(int)((nact + 255) / 256), 256, 0, stream>>>(x_d, xd_bf, nact);

  int nww = DIM * DIM;
  transpose_bf16_kernel<<<(nww + 255) / 256, 256, 0, stream>>>(Wq, WqT, DIM, DIM);
  transpose_bf16_kernel<<<(nww + 255) / 256, 256, 0, stream>>>(Wk, WkT, DIM, DIM);
  transpose_bf16_kernel<<<(nww + 255) / 256, 256, 0, stream>>>(Wv, WvT, DIM, DIM);
  int nw1 = DIM * MLP;
  transpose_bf16_kernel<<<(nw1 + 255) / 256, 256, 0, stream>>>(W1, W1T, DIM, MLP);
  transpose_bf16_kernel<<<(nw1 + 255) / 256, 256, 0, stream>>>(W2, W2T, MLP, DIM);

  dim3 gq(MROWS / 64, DIM / 256, 3);
  qkv_gemm_kernel<<<gq, 256, 0, stream>>>(xr_bf, xd_bf, WqT, WkT, WvT, Qb, Kb, Vt);

  attn_kernel<<<BATCH * 8 * (NSEQ / 16), 256, 0, stream>>>(Qb, Kb, Vt, attnO);

  ln_kernel<<<MROWS, 256, 0, stream>>>(attnO, x_r, ln_g, ln_b, fg, fb, out, hbf);

  dim3 g1(MROWS / 64, MLP / 256);
  ffn1_kernel<<<g1, 256, 0, stream>>>(hbf, W1T, b1, H1);

  dim3 g2(MROWS / 64, DIM / 256);
  ffn2_kernel<<<g2, 256, 0, stream>>>(H1, W2T, b2, out);
}
